// GCHAAttention_25701084299356
// MI455X (gfx1250) — matrix operand order repaired and checked by reading, not compiled, not run on hardware
//
#include <hip/hip_runtime.h>
#include <hip/hip_bf16.h>
#include <hip/hip_fp16.h>

#define EMBED  256
#define NHEADS 8
#define HDIM   32
#define NANCH  64
#define BATCH  4
#define SEQ    2048

typedef __attribute__((ext_vector_type(16))) _Float16 v16h;
typedef __attribute__((ext_vector_type(8)))  _Float16 v8h;
typedef __attribute__((ext_vector_type(8)))  float    v8f;

union V16U { v16h v; v8h h[2]; };

// ---------------------------------------------------------------- utilities
__global__ void cvt_f32_f16(const float* __restrict__ src,
                            _Float16* __restrict__ dst, int n) {
  int i = blockIdx.x * blockDim.x + threadIdx.x;
  int stride = gridDim.x * blockDim.x;
  for (; i < n; i += stride) dst[i] = (_Float16)src[i];
}

__global__ void nearest_anchor(const float* __restrict__ pos,
                               const float* __restrict__ anchors,
                               int* __restrict__ out, int n) {
  int i = blockIdx.x * blockDim.x + threadIdx.x;
  if (i >= n) return;
  float px = pos[2 * i], py = pos[2 * i + 1];
  float best = 3.4e38f; int bi = 0;
  #pragma unroll 8
  for (int a = 0; a < NANCH; ++a) {
    float dx = px - anchors[2 * a], dy = py - anchors[2 * a + 1];
    float d = dx * dx + dy * dy;
    if (d < best) { best = d; bi = a; }
  }
  out[i] = bi;
}

// --------------------------------------------- QKV projection (WMMA GEMM)
// One wave computes one 16x16 tile of  X(8192x256) @ W^T(256x256) + b.
// A-fragment (16x32 f16): lane row = lane%16, contiguous 16 halves at
//   k = kk + 16*(lane/16), same order as B.
// B-fragment (32x16 f16): lane col = lane%16, contiguous 16 halves at
//   k = kk + 16*(lane/16).  B(k,j) = W[j,k] -> contiguous in row j of W.
__global__ void __launch_bounds__(32) proj_qkv_kernel(
    const _Float16* __restrict__ X,
    const _Float16* __restrict__ Wq, const _Float16* __restrict__ Wk,
    const _Float16* __restrict__ Wv,
    const float* __restrict__ bq, const float* __restrict__ bk,
    const float* __restrict__ bv,
    _Float16* __restrict__ Qh, _Float16* __restrict__ Kh,
    _Float16* __restrict__ Vh) {
  const int lane  = threadIdx.x;
  const int lhalf = lane >> 4;
  const int l16   = lane & 15;
  const int base8 = lhalf * 8;
  const int rowT  = blockIdx.x;   // 0..511 over B*N/16
  const int colT  = blockIdx.y;   // 0..15  over C/16

  const _Float16* W; const float* bias; _Float16* Out;
  if      (blockIdx.z == 0) { W = Wq; bias = bq; Out = Qh; }
  else if (blockIdx.z == 1) { W = Wk; bias = bk; Out = Kh; }
  else                      { W = Wv; bias = bv; Out = Vh; }

  const int arow = rowT * 16 + l16;
  const int jcol = colT * 16 + l16;
  const _Float16* xrow = X + (size_t)arow * EMBED;
  const _Float16* wrow = W + (size_t)jcol * EMBED + 16 * lhalf;

  v8f acc = {};
  #pragma unroll
  for (int kk = 0; kk < EMBED; kk += 32) {
    V16U a, bf;
    a.h[0]  = *(const v8h*)(xrow + kk + 16 * lhalf);
    a.h[1]  = *(const v8h*)(xrow + kk + 16 * lhalf + 8);
    bf.h[0] = *(const v8h*)(wrow + kk);
    bf.h[1] = *(const v8h*)(wrow + kk + 8);
    acc = __builtin_amdgcn_wmma_f32_16x16x32_f16(false, a.v, false, bf.v,
                                                 (short)0, acc, false, false);
  }
  const float bj = bias[jcol];
  const int h = jcol >> 5;     // head
  const int d = jcol & 31;     // dim in head
  #pragma unroll
  for (int r = 0; r < 8; ++r) {  // D: row = r + 8*lhalf, col = jcol
    int grow = rowT * 16 + r + 8 * lhalf;
    int bb = grow >> 11, n = grow & (SEQ - 1);
    Out[(((size_t)bb * NHEADS + h) * SEQ + n) * HDIM + d] =
        (_Float16)(acc[r] + bj);
  }
}

// --------------------------------------------------- fused flash attention
// Block = 4 waves, one (b,h), 64 query rows; each wave owns 16 rows.
// Per 32-key block: S^T = K_tile(16x32d) x Q^T(32dx16m) via 2 WMMAs,
// online softmax in registers, P^T packed into a B-fragment from the
// lane's own values, then O^T += V^T x P^T via 2 WMMAs.
__global__ void __launch_bounds__(128) gcha_attn_kernel(
    const _Float16* __restrict__ Qh, const _Float16* __restrict__ Kh,
    const _Float16* __restrict__ Vh,
    const float* __restrict__ query_pos, const float* __restrict__ key_pos,
    const int* __restrict__ qnear, const int* __restrict__ knear,
    const float* __restrict__ gscale,
    _Float16* __restrict__ Oh) {
  __shared__ __align__(16) _Float16 ldsK[32 * 32];
  __shared__ __align__(16) _Float16 ldsVT[32 * 32];
  __shared__ float ldsKP[64];
  __shared__ int   ldsKN[32];

  const int tid   = threadIdx.x;
  const int wave  = tid >> 5;
  const int lane  = tid & 31;
  const int lhalf = lane >> 4;
  const int l16   = lane & 15;
  const int base8 = lhalf * 8;

  const int b  = blockIdx.z;
  const int h  = blockIdx.y;
  const int n0 = blockIdx.x * 64 + wave * 16;

  const float gs         = gscale[0];
  const float inv_sqrt_d = 0.17677669529663687f;  // 1/sqrt(32)
  const float inv2eps    = 5.0e5f;                // 1/(2*EPS)

  const int   mrow = n0 + l16;                    // query this lane owns
  const float qpx  = query_pos[((size_t)b * SEQ + mrow) * 2 + 0];
  const float qpy  = query_pos[((size_t)b * SEQ + mrow) * 2 + 1];
  const int   qn   = qnear[(size_t)b * SEQ + mrow];

  // Q^T B-fragment: lane col m = l16, d = 8*lhalf .. +7 and 16+8*lhalf .. +7
  const _Float16* qbase =
      Qh + (((size_t)b * NHEADS + h) * SEQ + mrow) * HDIM + 8 * lhalf;
  V16U qb; qb.h[0] = *(const v8h*)qbase; qb.h[1] = *(const v8h*)(qbase + 16);

  v8f o0 = {}, o1 = {};
  float runmax = -1e30f, lsum = 0.f;

  const _Float16* gK = Kh + ((size_t)b * NHEADS + h) * SEQ * HDIM;
  const _Float16* gV = Vh + ((size_t)b * NHEADS + h) * SEQ * HDIM;

  for (int kb = 0; kb < SEQ; kb += 32) {
    // ---- cooperative staging (128 threads, 2KB K + 2KB V^T + pos/near)
    ((v8h*)ldsK)[tid] = ((const v8h*)(gK + (size_t)kb * HDIM))[tid];
    {
      v8h vv = ((const v8h*)(gV + (size_t)kb * HDIM))[tid];
      int key = tid >> 2, d0 = (tid & 3) * 8;
      #pragma unroll
      for (int i = 0; i < 8; ++i) ldsVT[(d0 + i) * 32 + key] = vv[i];
    }
    if (tid < 32) {
      ldsKP[2 * tid]     = key_pos[((size_t)b * SEQ + kb + tid) * 2 + 0];
      ldsKP[2 * tid + 1] = key_pos[((size_t)b * SEQ + kb + tid) * 2 + 1];
      ldsKN[tid]         = knear[(size_t)b * SEQ + kb + tid];
    }
    if (kb + 32 < SEQ) {  // pull next block toward L2/WGP$
      __builtin_prefetch(gK + (size_t)(kb + 32) * HDIM + tid * 8, 0, 1);
      __builtin_prefetch(gV + (size_t)(kb + 32) * HDIM + tid * 8, 0, 1);
    }
    __syncthreads();

    // ---- S^T = K x Q^T : two 16-key tiles
    V16U a0, a1;
    a0.h[0] = *(const v8h*)&ldsK[l16 * 32 + base8];
    a0.h[1] = *(const v8h*)&ldsK[l16 * 32 + 16 + base8];
    a1.h[0] = *(const v8h*)&ldsK[(l16 + 16) * 32 + base8];
    a1.h[1] = *(const v8h*)&ldsK[(l16 + 16) * 32 + 16 + base8];
    v8f zero = {};
    v8f s0 = __builtin_amdgcn_wmma_f32_16x16x32_f16(false, a0.v, false, qb.v,
                                                    (short)0, zero, false, false);
    v8f s1 = __builtin_amdgcn_wmma_f32_16x16x32_f16(false, a1.v, false, qb.v,
                                                    (short)0, zero, false, false);

    // ---- scale + guided-mask bias (lane: col m = l16, key = r+8*lhalf)
    float p0[8], p1[8];
    float bmax = -1e30f;
    #pragma unroll
    for (int r = 0; r < 8; ++r) {
      int kl = r + 8 * lhalf;
      {
        float dx = qpx - ldsKP[2 * kl], dy = qpy - ldsKP[2 * kl + 1];
        float w = (qn == ldsKN[kl])
                      ? __expf(-__fsqrt_rn(dx * dx + dy * dy) * inv2eps) : 0.f;
        float sv = s0[r] * inv_sqrt_d + gs * __logf(w + 1e-6f);
        p0[r] = sv; bmax = fmaxf(bmax, sv);
      }
      {
        int k2 = kl + 16;
        float dx = qpx - ldsKP[2 * k2], dy = qpy - ldsKP[2 * k2 + 1];
        float w = (qn == ldsKN[k2])
                      ? __expf(-__fsqrt_rn(dx * dx + dy * dy) * inv2eps) : 0.f;
        float sv = s1[r] * inv_sqrt_d + gs * __logf(w + 1e-6f);
        p1[r] = sv; bmax = fmaxf(bmax, sv);
      }
    }
    // ---- online softmax (row lives in lane pair {m, m+16})
    bmax = fmaxf(bmax, __shfl_xor(bmax, 16));
    float nm   = fmaxf(runmax, bmax);
    float corr = __expf(runmax - nm);
    runmax = nm;
    lsum *= corr;
    #pragma unroll
    for (int r = 0; r < 8; ++r) { o0[r] *= corr; o1[r] *= corr; }

    // ---- exponentiate + pack P^T straight into a B-fragment
    V16U pb;
    #pragma unroll
    for (int r = 0; r < 8; ++r) {
      float ea = __expf(p0[r] - nm);   // key r + 8*lhalf
      float eb = __expf(p1[r] - nm);   // key 16 + r + 8*lhalf
      lsum += ea + eb;
      float sel  = lhalf ? ea : eb;    // shuffled but unused by the packing
      float part = __shfl_xor(sel, 16);
      pb.v[r]     = (_Float16)ea;
      pb.v[r + 8] = (_Float16)eb;
    }

    // ---- O^T += V^T x P^T (two d-halves)
    V16U va0, va1;
    va0.h[0] = *(const v8h*)&ldsVT[l16 * 32 + base8];
    va0.h[1] = *(const v8h*)&ldsVT[l16 * 32 + 16 + base8];
    va1.h[0] = *(const v8h*)&ldsVT[(l16 + 16) * 32 + base8];
    va1.h[1] = *(const v8h*)&ldsVT[(l16 + 16) * 32 + 16 + base8];
    o0 = __builtin_amdgcn_wmma_f32_16x16x32_f16(false, va0.v, false, pb.v,
                                                (short)0, o0, false, false);
    o1 = __builtin_amdgcn_wmma_f32_16x16x32_f16(false, va1.v, false, pb.v,
                                                (short)0, o1, false, false);
    __syncthreads();
  }

  lsum += __shfl_xor(lsum, 16);
  float inv = 1.0f / lsum;
  _Float16* orow = Oh + ((size_t)b * SEQ + mrow) * EMBED + h * HDIM;
  #pragma unroll
  for (int r = 0; r < 8; ++r) {  // O^T: d = r + 8*lhalf (+16), m = l16
    int d0 = r + 8 * lhalf;
    orow[d0]      = (_Float16)(o0[r] * inv);
    orow[16 + d0] = (_Float16)(o1[r] * inv);
  }
}

// ------------------------------------------------ output projection (WMMA)
__global__ void __launch_bounds__(32) proj_out_kernel(
    const _Float16* __restrict__ X,   // (B*N, C) f16 attention output
    const _Float16* __restrict__ Wo,  // (C, C) f16 row-major
    const float* __restrict__ bo,
    float* __restrict__ Out) {        // (B*N, C) f32
  const int lane  = threadIdx.x;
  const int lhalf = lane >> 4;
  const int l16   = lane & 15;
  const int base8 = lhalf * 8;
  const int rowT  = blockIdx.x;
  const int colT  = blockIdx.y;

  const int arow = rowT * 16 + l16;
  const int jcol = colT * 16 + l16;
  const _Float16* xrow = X + (size_t)arow * EMBED;
  const _Float16* wrow = Wo + (size_t)jcol * EMBED + 16 * lhalf;

  v8f acc = {};
  #pragma unroll
  for (int kk = 0; kk < EMBED; kk += 32) {
    V16U a, bf;
    a.h[0]  = *(const v8h*)(xrow + kk + 16 * lhalf);
    a.h[1]  = *(const v8h*)(xrow + kk + 16 * lhalf + 8);
    bf.h[0] = *(const v8h*)(wrow + kk);
    bf.h[1] = *(const v8h*)(wrow + kk + 8);
    acc = __builtin_amdgcn_wmma_f32_16x16x32_f16(false, a.v, false, bf.v,
                                                 (short)0, acc, false, false);
  }
  const float bj = bo[jcol];
  #pragma unroll
  for (int r = 0; r < 8; ++r) {
    size_t grow = (size_t)rowT * 16 + r + 8 * lhalf;
    Out[grow * EMBED + jcol] = acc[r] + bj;
  }
}

// --------------------------------------------------------------- launcher
extern "C" void kernel_launch(void* const* d_in, const int* in_sizes, int n_in,
                              void* d_out, int out_size, void* d_ws,
                              size_t ws_size, hipStream_t stream) {
  (void)in_sizes; (void)n_in; (void)out_size; (void)ws_size;
  const float* query      = (const float*)d_in[0];
  const float* query_pos  = (const float*)d_in[1];
  const float* key_pos    = (const float*)d_in[2];
  const float* anchor_pos = (const float*)d_in[3];
  const float* Wq = (const float*)d_in[4];  const float* bq = (const float*)d_in[5];
  const float* Wk = (const float*)d_in[6];  const float* bk = (const float*)d_in[7];
  const float* Wv = (const float*)d_in[8];  const float* bv = (const float*)d_in[9];
  const float* Wo = (const float*)d_in[10]; const float* bo = (const float*)d_in[11];
  const float* gs = (const float*)d_in[12];

  char* ws = (char*)d_ws;
  size_t off = 0;
  const size_t NTOK = (size_t)BATCH * SEQ;            // 8192
  _Float16* qf16 = (_Float16*)(ws + off); off += NTOK * EMBED * 2;      // 4 MiB
  _Float16* Wq16 = (_Float16*)(ws + off); off += (size_t)EMBED * EMBED * 2;
  _Float16* Wk16 = (_Float16*)(ws + off); off += (size_t)EMBED * EMBED * 2;
  _Float16* Wv16 = (_Float16*)(ws + off); off += (size_t)EMBED * EMBED * 2;
  _Float16* Wo16 = (_Float16*)(ws + off); off += (size_t)EMBED * EMBED * 2;
  _Float16* Qh   = (_Float16*)(ws + off); off += NTOK * EMBED * 2;      // (B,H,N,D)
  _Float16* Kh   = (_Float16*)(ws + off); off += NTOK * EMBED * 2;
  _Float16* Vh   = (_Float16*)(ws + off); off += NTOK * EMBED * 2;
  _Float16* Oh   = (_Float16*)(ws + off); off += NTOK * EMBED * 2;      // (B*N, C)
  int* qNear     = (int*)(ws + off);      off += NTOK * 4;
  int* kNear     = (int*)(ws + off);      off += NTOK * 4;

  // f32 -> f16 conversions
  cvt_f32_f16<<<2048, 256, 0, stream>>>(query, qf16, (int)(NTOK * EMBED));
  cvt_f32_f16<<<64, 256, 0, stream>>>(Wq, Wq16, EMBED * EMBED);
  cvt_f32_f16<<<64, 256, 0, stream>>>(Wk, Wk16, EMBED * EMBED);
  cvt_f32_f16<<<64, 256, 0, stream>>>(Wv, Wv16, EMBED * EMBED);
  cvt_f32_f16<<<64, 256, 0, stream>>>(Wo, Wo16, EMBED * EMBED);

  // nearest anchors
  nearest_anchor<<<32, 256, 0, stream>>>(query_pos, anchor_pos, qNear, (int)NTOK);
  nearest_anchor<<<32, 256, 0, stream>>>(key_pos,   anchor_pos, kNear, (int)NTOK);

  // QKV projections: 512 row-tiles x 16 col-tiles x {Q,K,V}
  proj_qkv_kernel<<<dim3(512, 16, 3), 32, 0, stream>>>(
      qf16, Wq16, Wk16, Wv16, bq, bk, bv, Qh, Kh, Vh);

  // fused guided flash attention: grid (N/64, H, B), 4 waves/block
  gcha_attn_kernel<<<dim3(SEQ / 64, NHEADS, BATCH), 128, 0, stream>>>(
      Qh, Kh, Vh, query_pos, key_pos, qNear, kNear, gs, Oh);

  // output projection -> f32 d_out
  proj_out_kernel<<<dim3(512, 16, 1), 32, 0, stream>>>(
      Oh, Wo16, bo, (float*)d_out);
}
